// MaxSim_24996709663181
// MI455X (gfx1250) — matrix operand order repaired and checked by reading, not compiled, not run on hardware
//
#include <hip/hip_runtime.h>
#include <hip/hip_bf16.h>

// Problem dims (from reference): B=32, Lq=64, Lk=256, D=128, N=32 negatives.
#define B_    32
#define LQ    64
#define LK    256
#define D_    128
#define NNEG  32
#define NCLS  33            // 1 positive + 32 negatives
#define KSTR  136           // LDS row stride in bf16 elems (128 + 8 pad -> bank-conflict-free)
#define INV_T 20.0f         // 1 / temperature(0.05)

typedef __attribute__((ext_vector_type(16))) __bf16 v16bf;
typedef __attribute__((ext_vector_type(8)))  float  v8f;

union BF16x16 { v16bf v; unsigned u[8]; };

__device__ __forceinline__ unsigned short f2bf(float f) {
    // round-to-nearest-even f32 -> bf16 bits
    unsigned u = __builtin_bit_cast(unsigned, f);
    u += 0x7fffu + ((u >> 16) & 1u);
    return (unsigned short)(u >> 16);
}

// One block per (batch b, class j): j==0 -> positive key tile, j>=1 -> negative j-1.
// Computes logits[b][j] = sum_q (1/||q||) * max_k [ (q . k) * (1/||k||) ]
__global__ __launch_bounds__(256)
void maxsim_scores(const float* __restrict__ q,
                   const float* __restrict__ pk,
                   const float* __restrict__ nk,
                   float* __restrict__ logits)
{
    __shared__ unsigned short sQ[LQ * KSTR];   // q rows as bf16
    __shared__ unsigned short sK[LK * KSTR];   // key rows as bf16
    __shared__ float sQInv[LQ];                // 1 / max(||q_row||, eps)
    __shared__ float sKInv[LK];                // 1 / max(||k_row||, eps)
    __shared__ float sPart[2][LQ];             // per-nhalf row maxima
    __shared__ float sRed[LQ];

    const int bid = blockIdx.x;
    const int b = bid / NCLS;
    const int j = bid % NCLS;

    const float* key = (j == 0)
        ? (pk + (size_t)b * LK * D_)
        : (nk + (((size_t)b * NNEG) + (size_t)(j - 1)) * (size_t)(LK * D_));
    const float* qb = q + (size_t)b * LQ * D_;

    const int wave = threadIdx.x >> 5;
    const int lane = threadIdx.x & 31;

    // ---- stage keys: wave w loads rows [w*32, w*32+32); one coalesced row per iter ----
    for (int rr = 0; rr < 32; ++rr) {
        const int r = wave * 32 + rr;
        float4 kv = *(const float4*)(key + (size_t)r * D_ + lane * 4);
        float ss = kv.x*kv.x + kv.y*kv.y + kv.z*kv.z + kv.w*kv.w;
        #pragma unroll
        for (int off = 16; off > 0; off >>= 1) ss += __shfl_xor(ss, off, 32);
        unsigned lo = (unsigned)f2bf(kv.x) | ((unsigned)f2bf(kv.y) << 16);
        unsigned hi = (unsigned)f2bf(kv.z) | ((unsigned)f2bf(kv.w) << 16);
        *(uint2*)&sK[r * KSTR + lane * 4] = make_uint2(lo, hi);
        if (lane == 0) sKInv[r] = 1.0f / fmaxf(sqrtf(ss), 1e-12f);
    }
    // ---- stage q: wave w loads rows [w*8, w*8+8) ----
    for (int rr = 0; rr < 8; ++rr) {
        const int r = wave * 8 + rr;
        float4 qv = *(const float4*)(qb + (size_t)r * D_ + lane * 4);
        float ss = qv.x*qv.x + qv.y*qv.y + qv.z*qv.z + qv.w*qv.w;
        #pragma unroll
        for (int off = 16; off > 0; off >>= 1) ss += __shfl_xor(ss, off, 32);
        unsigned lo = (unsigned)f2bf(qv.x) | ((unsigned)f2bf(qv.y) << 16);
        unsigned hi = (unsigned)f2bf(qv.z) | ((unsigned)f2bf(qv.w) << 16);
        *(uint2*)&sQ[r * KSTR + lane * 4] = make_uint2(lo, hi);
        if (lane == 0) sQInv[r] = 1.0f / fmaxf(sqrtf(ss), 1e-12f);
    }
    __syncthreads();

    // ---- compute: 8 waves -> (mtile 0..3) x (nhalf 0..1); each wave: 8 N-tiles ----
    const int mtile = wave >> 1;
    const int nhalf = wave & 1;
    const int l16 = lane & 15;

    // A fragments (16x32 bf16, ISA 7.12.2): lane M = l16;
    // VGPR v, half h(=lane>=16): K = (v<4?0:16) + (v&3)*2 + h*8  (+ ks*32)
    BF16x16 afrag[4];
    {
        const int arow = (mtile * 16 + l16) * KSTR;
        const int kbase = (lane < 16) ? 0 : 8;
        #pragma unroll
        for (int ks = 0; ks < 4; ++ks) {
            #pragma unroll
            for (int v = 0; v < 8; ++v) {
                const int k = ks * 32 + ((v < 4) ? 0 : 16) + (v & 3) * 2 + kbase;
                afrag[ks].u[v] = *(const unsigned*)&sQ[arow + k];
            }
        }
    }

    v8f rowmax;
    #pragma unroll
    for (int i = 0; i < 8; ++i) rowmax[i] = -3.402823466e38f;

    // B fragments (32x16 bf16): lane N = l16; VGPR v, half h: K = (v<4?0:16) + (v&3)*2 + h*8 (+ ks*32)
    const int kbB = (lane < 16) ? 0 : 8;
    #pragma unroll
    for (int nt = 0; nt < 8; ++nt) {
        const int col = (nhalf * 8 + nt) * 16 + l16;   // key row index (column of S)
        const int brow = col * KSTR;
        const float cs = sKInv[col];                    // per-column 1/||k||, per-lane const
        v8f acc = {};
        #pragma unroll
        for (int ks = 0; ks < 4; ++ks) {
            BF16x16 bfrag;
            #pragma unroll
            for (int v = 0; v < 8; ++v)
                bfrag.u[v] = *(const unsigned*)&sK[brow + ks * 32 + kbB + ((v < 4) ? 0 : 16) + (v & 3) * 2];
            acc = __builtin_amdgcn_wmma_f32_16x16x32_bf16(
                      false, afrag[ks].v, false, bfrag.v, (short)0, acc, false, false);
        }
        #pragma unroll
        for (int i = 0; i < 8; ++i) rowmax[i] = fmaxf(rowmax[i], acc[i] * cs);
    }

    // max over the 16 N-lanes of each half (xor 8/4/2/1 stays within halves)
    #pragma unroll
    for (int off = 8; off > 0; off >>= 1) {
        #pragma unroll
        for (int i = 0; i < 8; ++i)
            rowmax[i] = fmaxf(rowmax[i], __shfl_xor(rowmax[i], off, 32));
    }

    if (l16 == 0) {
        const int half = lane >> 4;   // D layout: VGPR i holds row i (lanes<16) / i+8
        #pragma unroll
        for (int i = 0; i < 8; ++i)
            sPart[nhalf][mtile * 16 + half * 8 + i] = rowmax[i];
    }
    __syncthreads();

    if (threadIdx.x < LQ) {
        const int t = threadIdx.x;
        const float m = fmaxf(sPart[0][t], sPart[1][t]);
        sRed[t] = sQInv[t] * m;
    }
    __syncthreads();
    if (threadIdx.x == 0) {
        float s = 0.0f;
        for (int t = 0; t < LQ; ++t) s += sRed[t];
        logits[b * NCLS + j] = s;
    }
}

// Final 32x33 log-softmax cross-entropy (labels==0), mean over batch. One wave.
__global__ void maxsim_loss(const float* __restrict__ logits, float* __restrict__ out)
{
    const int t = threadIdx.x;          // 32 threads, one per batch row
    const float* row = logits + t * NCLS;
    const float x0 = row[0] * INV_T;
    float mx = x0;
    for (int jj = 1; jj < NCLS; ++jj) mx = fmaxf(mx, row[jj] * INV_T);
    float se = 0.0f;
    for (int jj = 0; jj < NCLS; ++jj) se += expf(row[jj] * INV_T - mx);
    float loss = (mx + logf(se)) - x0;
    #pragma unroll
    for (int off = 16; off > 0; off >>= 1) loss += __shfl_xor(loss, off, 32);
    if (t == 0) out[0] = loss * (1.0f / (float)B_);
}

extern "C" void kernel_launch(void* const* d_in, const int* in_sizes, int n_in,
                              void* d_out, int out_size, void* d_ws, size_t ws_size,
                              hipStream_t stream)
{
    (void)in_sizes; (void)n_in; (void)out_size; (void)ws_size;
    const float* q  = (const float*)d_in[0];   // query      [32,64,128]
    const float* pk = (const float*)d_in[2];   // pos_key    [32,256,128]
    const float* nk = (const float*)d_in[4];   // neg_key    [32,32,256,128]
    // masks (d_in[1], d_in[3], d_in[5]) are unused by the reference
    float* logits = (float*)d_ws;              // [32][33] f32 scratch

    maxsim_scores<<<B_ * NCLS, 256, 0, stream>>>(q, pk, nk, logits);
    maxsim_loss<<<1, 32, 0, stream>>>(logits, (float*)d_out);
}
